// BiLSTM_6176162972060
// MI455X (gfx1250) — matrix operand order repaired and checked by reading, not compiled, not run on hardware
//
#include <hip/hip_runtime.h>
#include <hip/hip_bf16.h>
#include <stdint.h>

// ---------------------------------------------------------------------------
// BiLSTM on gfx1250: bf16 WMMA (16x16x32, f32 accum) for both the hoisted
// input-projection GEMM and the per-step recurrent GEMM. Persistent scan
// kernel with one device-scope barrier per timestep; cell state in VGPRs.
// ---------------------------------------------------------------------------

typedef __bf16 bf16_t;
typedef __attribute__((ext_vector_type(16))) __bf16 v16bf;
typedef __attribute__((ext_vector_type(8)))  __bf16 v8bf;
typedef __attribute__((ext_vector_type(8)))  float   v8f;

#define T_STEPS 512
#define BATCH   128
#define HID     400
#define GATES   1600                 // 4*HID
#define M_TOT   (T_STEPS * BATCH)    // 65536 rows
#define KPAD0   416                  // pad 400 -> multiple of 32
#define KPAD12  832                  // pad 800 -> multiple of 32
#define KPADH   416
#define NSLICES 25                   // HID / 16
#define SCAN_WGS (2 * NSLICES)       // 2 directions x 25 col-slices

// ---- WMMA tile loaders (layouts per CDNA5 ISA 7.12.2) ----------
// A matrix 16x32 bf16, M on lanes&15; K split [0..7|16..23] / [8..15|24..31]
__device__ inline v16bf load_tile_A(const bf16_t* __restrict__ p, int ld,
                                    int row0, int k0, int half, int lid) {
  const bf16_t* base = p + (size_t)(row0 + lid) * ld + k0 + half * 8;
  v8bf lo = *(const v8bf*)(base);        // K = half*8 .. half*8+7
  v8bf hi = *(const v8bf*)(base + 16);   // K = 16+half*8 ..
  v16bf r;
#pragma unroll
  for (int i = 0; i < 8; ++i) { r[i] = lo[i]; r[i + 8] = hi[i]; }
  return r;
}
// B matrix 32x16 bf16 stored as W^T (N-major, K contiguous):
// K split [0..7|16..23] / [8..15|24..31] (same as A) for column N=lane&15.
__device__ inline v16bf load_tile_B(const bf16_t* __restrict__ p, int ld,
                                    int row0, int k0, int half, int lid) {
  const bf16_t* base = p + (size_t)(row0 + lid) * ld + k0 + half * 8;
  v8bf lo = *(const v8bf*)(base);
  v8bf hi = *(const v8bf*)(base + 16);
  v16bf r;
#pragma unroll
  for (int i = 0; i < 8; ++i) { r[i] = lo[i]; r[i + 8] = hi[i]; }
  return r;
}

// ---- fp32 -> bf16 convert with zero K-padding -----------------------------
__global__ void cvt_pad_bf16(const float* __restrict__ src, bf16_t* __restrict__ dst,
                             int rows, int K, int Kpad) {
  size_t total = (size_t)rows * Kpad;
  for (size_t idx = (size_t)blockIdx.x * blockDim.x + threadIdx.x; idx < total;
       idx += (size_t)gridDim.x * blockDim.x) {
    int col = (int)(idx % Kpad);
    size_t row = idx / Kpad;
    float v = (col < K) ? src[row * (size_t)K + col] : 0.0f;
    dst[idx] = (bf16_t)v;
  }
}

// Zero the K-pad columns [800,832) of an activation buffer (row stride 832).
__global__ void zero_pad_cols(bf16_t* __restrict__ act) {
  size_t total = (size_t)M_TOT * 32;
  for (size_t idx = (size_t)blockIdx.x * blockDim.x + threadIdx.x; idx < total;
       idx += (size_t)gridDim.x * blockDim.x) {
    size_t row = idx >> 5;
    int col = 800 + (int)(idx & 31);
    act[row * KPAD12 + col] = (bf16_t)0.0f;
  }
}

// ---- gx = X @ Wih^T + bias  (M=65536, N=1600, K in {416,832}) -------------
// One wave computes a 16(M) x 80(N) tile: A reused across 5 WMMA calls.
__global__ void __launch_bounds__(256)
gx_gemm(const bf16_t* __restrict__ X, const bf16_t* __restrict__ W,
        const float* __restrict__ bias, float* __restrict__ GX, int Kpad) {
  const int lane = threadIdx.x & 31;
  const int half = lane >> 4, lid = lane & 15;
  const int waveId = blockIdx.x * (blockDim.x >> 5) + (threadIdx.x >> 5);
  const int NT = GATES / 80;  // 20 column-tasks
  const int mt = waveId / NT;
  const int nt = waveId % NT;
  if (mt >= M_TOT / 16) return;
  const int m0 = mt * 16;
  const int n0 = nt * 80;

  const v8f zero = {0.f, 0.f, 0.f, 0.f, 0.f, 0.f, 0.f, 0.f};
  v8f acc[5];
#pragma unroll
  for (int j = 0; j < 5; ++j) acc[j] = zero;

  for (int k0 = 0; k0 < Kpad; k0 += 32) {
    __builtin_prefetch(X + (size_t)(m0 + lid) * Kpad + k0 + 128, 0, 0);
    v16bf a = load_tile_A(X, Kpad, m0, k0, half, lid);
#pragma unroll
    for (int j = 0; j < 5; ++j) {
      v16bf b = load_tile_B(W, Kpad, n0 + j * 16, k0, half, lid);
      acc[j] = __builtin_amdgcn_wmma_f32_16x16x32_bf16(
          false, a, false, b, (short)0, acc[j], false, false);
    }
  }
#pragma unroll
  for (int j = 0; j < 5; ++j) {
    const int col = n0 + j * 16 + lid;
    const float bv = bias[col];
#pragma unroll
    for (int vr = 0; vr < 8; ++vr) {
      int row = m0 + vr + half * 8;
      GX[(size_t)row * GATES + col] = acc[j][vr] + bv;
    }
  }
}

// ---- persistent bidirectional LSTM scan -----------------------------------
// grid = 50 WGs x 256 threads. dir = blk/25, slice = blk%25, wave = M-tile.
// Each wave owns a 16(batch) x 16(h) block: c and carried h stay in VGPRs.
__global__ void __launch_bounds__(256)
lstm_scan(const float* __restrict__ GX,        // [2][M_TOT][1600] fp32
          const bf16_t* __restrict__ WhhF,     // [1600][416] bf16 (N-major)
          const bf16_t* __restrict__ WhhB,
          const int* __restrict__ lengths,     // [128]
          bf16_t* __restrict__ hbuf,           // [2 dir][2 pp][128][416] bf16
          bf16_t* __restrict__ actout,         // [M_TOT][832] bf16
          int* __restrict__ counters) {
  const int lane = threadIdx.x & 31;
  const int half = lane >> 4, lid = lane & 15;
  const int wave = threadIdx.x >> 5;
  const int dir = blockIdx.x / NSLICES;
  const int slice = blockIdx.x % NSLICES;

  const bf16_t* Whh = dir ? WhhB : WhhF;
  const float* gxd = GX + (size_t)dir * M_TOT * GATES;
  bf16_t* hb = hbuf + (size_t)dir * 2 * BATCH * KPADH;
  int* cnt = counters + dir * 16;

  const int m0 = wave * 16;        // batch rows m0..m0+15
  const int n0c = slice * 16;      // h columns n0c..n0c+15
  const int col = n0c + lid;

  float c_state[8], h_state[8];
  int len[8];
#pragma unroll
  for (int vr = 0; vr < 8; ++vr) {
    c_state[vr] = 0.f;
    h_state[vr] = 0.f;
    len[vr] = lengths[m0 + vr + half * 8];
  }

  for (int step = 0; step < T_STEPS; ++step) {
    const int t = dir ? (T_STEPS - 1 - step) : step;
    const bf16_t* hread = hb + (size_t)(step & 1) * BATCH * KPADH;
    bf16_t* hwrite = hb + (size_t)((step + 1) & 1) * BATCH * KPADH;

    // init accumulators from precomputed gx[t]
    v8f acc[4];
    const float* gxt = gxd + (size_t)t * BATCH * GATES;
#pragma unroll
    for (int g = 0; g < 4; ++g) {
      const int gc = g * HID + col;
#pragma unroll
      for (int vr = 0; vr < 8; ++vr) {
        int row = m0 + vr + half * 8;
        acc[g][vr] = gxt[(size_t)row * GATES + gc];
      }
    }

    // recurrent GEMM: + h_prev @ Whh^T  (A tile reused across 4 gates)
    for (int k0 = 0; k0 < KPADH; k0 += 32) {
      v16bf a = load_tile_A(hread, KPADH, m0, k0, half, lid);
#pragma unroll
      for (int g = 0; g < 4; ++g) {
        v16bf b = load_tile_B(Whh, KPADH, g * HID + n0c, k0, half, lid);
        acc[g] = __builtin_amdgcn_wmma_f32_16x16x32_bf16(
            false, a, false, b, (short)0, acc[g], false, false);
      }
    }

    // LSTM cell update (i, f, g, o order), masked by sequence length
#pragma unroll
    for (int vr = 0; vr < 8; ++vr) {
      const int row = m0 + vr + half * 8;
      const bool m = (t < len[vr]);
      float ig = 1.f / (1.f + __expf(-acc[0][vr]));
      float fg = 1.f / (1.f + __expf(-acc[1][vr]));
      float gg = tanhf(acc[2][vr]);
      float og = 1.f / (1.f + __expf(-acc[3][vr]));
      float cn = fg * c_state[vr] + ig * gg;
      float hn = og * tanhf(cn);
      float hc = m ? hn : h_state[vr];
      c_state[vr] = m ? cn : c_state[vr];
      h_state[vr] = hc;
      hwrite[(size_t)row * KPADH + col] = (bf16_t)hc;        // carried h
      actout[(size_t)(t * BATCH + row) * KPAD12 + dir * HID + col] =
          (bf16_t)(m ? hn : 0.f);                            // masked output
    }

    // device-scope barrier: publish hwrite before anyone reads next step
    __syncthreads();
    __threadfence();
    if (threadIdx.x == 0) {
      atomicAdd(cnt, 1);
      const int target = NSLICES * (step + 1);
      while (atomicAdd(cnt, 0) < target) { __builtin_amdgcn_s_sleep(1); }
    }
    __syncthreads();
  }
}

// ---- final bf16 activations -> fp32 output (T,B,800) ----------------------
__global__ void cvt_out(const bf16_t* __restrict__ act, float* __restrict__ out) {
  size_t total = (size_t)M_TOT * 800;
  for (size_t idx = (size_t)blockIdx.x * blockDim.x + threadIdx.x; idx < total;
       idx += (size_t)gridDim.x * blockDim.x) {
    size_t row = idx / 800;
    int c = (int)(idx % 800);
    out[idx] = (float)act[row * KPAD12 + c];
  }
}

// ---------------------------------------------------------------------------
extern "C" void kernel_launch(void* const* d_in, const int* in_sizes, int n_in,
                              void* d_out, int out_size, void* d_ws, size_t ws_size,
                              hipStream_t stream) {
  (void)in_sizes; (void)n_in; (void)out_size; (void)ws_size;

  const float* x       = (const float*)d_in[0];
  const int*   lengths = (const int*)d_in[1];
  const float* Wih_f0  = (const float*)d_in[2];
  const float* Wih_f12 = (const float*)d_in[3];
  const float* Whh_f   = (const float*)d_in[4];
  const float* b_f     = (const float*)d_in[5];
  const float* Wih_b0  = (const float*)d_in[6];
  const float* Wih_b12 = (const float*)d_in[7];
  const float* Whh_b   = (const float*)d_in[8];
  const float* b_b     = (const float*)d_in[9];
  float* out = (float*)d_out;

  // workspace carving
  uint8_t* p = (uint8_t*)d_ws;
  auto take = [&](size_t bytes) -> void* {
    void* r = (void*)p;
    p += (bytes + 255) & ~(size_t)255;
    return r;
  };
  bf16_t* x0bf = (bf16_t*)take((size_t)M_TOT * KPAD0 * 2);
  bf16_t* actA = (bf16_t*)take((size_t)M_TOT * KPAD12 * 2);
  bf16_t* actB = (bf16_t*)take((size_t)M_TOT * KPAD12 * 2);
  float*  gx   = (float*)take((size_t)2 * M_TOT * GATES * 4);
  bf16_t* wihF[3], *wihB[3], *whhF[3], *whhB[3];
  wihF[0] = (bf16_t*)take((size_t)GATES * KPAD0 * 2);
  wihF[1] = (bf16_t*)take((size_t)GATES * KPAD12 * 2);
  wihF[2] = (bf16_t*)take((size_t)GATES * KPAD12 * 2);
  wihB[0] = (bf16_t*)take((size_t)GATES * KPAD0 * 2);
  wihB[1] = (bf16_t*)take((size_t)GATES * KPAD12 * 2);
  wihB[2] = (bf16_t*)take((size_t)GATES * KPAD12 * 2);
  for (int l = 0; l < 3; ++l) whhF[l] = (bf16_t*)take((size_t)GATES * KPADH * 2);
  for (int l = 0; l < 3; ++l) whhB[l] = (bf16_t*)take((size_t)GATES * KPADH * 2);
  bf16_t* hbuf = (bf16_t*)take((size_t)2 * 2 * BATCH * KPADH * 2);
  int* counters = (int*)take(256);
  const size_t hbytes = (size_t)2 * 2 * BATCH * KPADH * 2;

  // bf16 conversions (weights + layer-0 input), zero K-padded
  cvt_pad_bf16<<<4096, 256, 0, stream>>>(x, x0bf, M_TOT, 400, KPAD0);
  cvt_pad_bf16<<<512, 256, 0, stream>>>(Wih_f0, wihF[0], GATES, 400, KPAD0);
  cvt_pad_bf16<<<512, 256, 0, stream>>>(Wih_b0, wihB[0], GATES, 400, KPAD0);
  for (int l = 1; l < 3; ++l) {
    cvt_pad_bf16<<<512, 256, 0, stream>>>(Wih_f12 + (size_t)(l - 1) * GATES * 800,
                                          wihF[l], GATES, 800, KPAD12);
    cvt_pad_bf16<<<512, 256, 0, stream>>>(Wih_b12 + (size_t)(l - 1) * GATES * 800,
                                          wihB[l], GATES, 800, KPAD12);
  }
  for (int l = 0; l < 3; ++l) {
    cvt_pad_bf16<<<512, 256, 0, stream>>>(Whh_f + (size_t)l * GATES * HID,
                                          whhF[l], GATES, HID, KPADH);
    cvt_pad_bf16<<<512, 256, 0, stream>>>(Whh_b + (size_t)l * GATES * HID,
                                          whhB[l], GATES, HID, KPADH);
  }

  const int gx_blocks = (M_TOT / 16) * (GATES / 80) / 8;  // 10240 (8 waves/WG)

  for (int l = 0; l < 3; ++l) {
    const bf16_t* in_act = (l == 0) ? x0bf : ((l == 1) ? actA : actB);
    bf16_t* out_act      = (l == 0) ? actA : ((l == 1) ? actB : actA);
    const int Kpad       = (l == 0) ? KPAD0 : KPAD12;

    if (l > 0)  // scrub K-pad columns of the previous layer's output
      zero_pad_cols<<<2048, 256, 0, stream>>>(const_cast<bf16_t*>(in_act));

    // hoisted input projections, both directions
    gx_gemm<<<gx_blocks, 256, 0, stream>>>(in_act, wihF[l], b_f + l * GATES,
                                           gx, Kpad);
    gx_gemm<<<gx_blocks, 256, 0, stream>>>(in_act, wihB[l], b_b + l * GATES,
                                           gx + (size_t)M_TOT * GATES, Kpad);

    hipMemsetAsync(hbuf, 0, hbytes, stream);      // h0 = 0 (+ zero K-pads)
    hipMemsetAsync(counters, 0, 256, stream);     // reset step barrier

    lstm_scan<<<SCAN_WGS, 256, 0, stream>>>(gx, whhF[l], whhB[l], lengths,
                                            hbuf, out_act, counters);
  }

  cvt_out<<<8192, 256, 0, stream>>>(actA, out);   // layer-2 output lives in actA
}
